// GegenbauerKANLayer_5299989643385
// MI455X (gfx1250) — matrix operand order repaired and checked by reading, not compiled, not run on hardware
//
#include <hip/hip_runtime.h>
#include <hip/hip_bf16.h>

#define IN_DIMS  512
#define OUT_DIMS 512
#define DEG      7
#define NPOLY    8                      // DEGREE+1
#define KDIM     (IN_DIMS * NPOLY)      // 4096
#define BATCH    16384

typedef __attribute__((ext_vector_type(16))) _Float16 v16h;
typedef __attribute__((ext_vector_type(8)))  _Float16 v8h;
typedef __attribute__((ext_vector_type(8)))  float    v8f;

// ---------------------------------------------------------------------------
// Pack coeffs (I, O, D+1) f32  ->  B (K=I*8, O) f16, k = i*8 + d  (K-major)
// ---------------------------------------------------------------------------
__global__ __launch_bounds__(256) void geg_pack_coeffs(const float* __restrict__ coeffs,
                                                       _Float16* __restrict__ Bh) {
    int idx = blockIdx.x * 256 + threadIdx.x;        // over KDIM * OUT_DIMS
    int k = idx / OUT_DIMS;
    int o = idx - k * OUT_DIMS;
    int i = k >> 3;
    int d = k & 7;
    Bh[idx] = (_Float16)coeffs[((size_t)i * OUT_DIMS + o) * NPOLY + d];
}

// ---------------------------------------------------------------------------
// A[b, i*8+d] = C_d(tanh(x[b,i])) in f16. One thread per (b,i): 16B store.
// alpha = 1:  C_{n+1} = (2(n+1) t C_n - (n+1) C_{n-1}) / (n+1)
// ---------------------------------------------------------------------------
__global__ __launch_bounds__(256) void geg_poly(const float* __restrict__ x,
                                                _Float16* __restrict__ Ah) {
    int idx = blockIdx.x * 256 + threadIdx.x;        // over chunk * IN_DIMS
    float t = tanhf(x[idx]);
    float p[NPOLY];
    p[0] = 1.0f;
    p[1] = 2.0f * t;                                 // 2*alpha*x
    #pragma unroll
    for (int n = 1; n < DEG; ++n) {
        float np1 = (float)(n + 1);
        p[n + 1] = (2.0f * np1 * t * p[n] - np1 * p[n - 1]) / np1;
    }
    v8h o;
    #pragma unroll
    for (int d = 0; d < NPOLY; ++d) o[d] = (_Float16)p[d];
    *(v8h*)(Ah + (size_t)idx * NPOLY) = o;
}

// ---------------------------------------------------------------------------
// GEMM: C(rows x 512) = A(rows x 4096, f16 row-major) @ B(4096 x 512, f16)
// 8 waves/block, wave tile 32x64 (2x4 WMMA 16x16x32_f16 tiles), block 128x128.
// ---------------------------------------------------------------------------
__global__ __launch_bounds__(256) void geg_gemm(const _Float16* __restrict__ A,
                                                const _Float16* __restrict__ B,
                                                float* __restrict__ C) {
    const int lane   = threadIdx.x & 31;
    const int wave   = threadIdx.x >> 5;
    const int waveM  = wave & 3;                     // 4 waves along M
    const int waveN  = wave >> 2;                    // 2 waves along N
    const int rowBase = blockIdx.x * 128 + waveM * 32;
    const int colBase = blockIdx.y * 128 + waveN * 64;
    const int mLane   = lane & 15;
    const int halfSel = lane >> 4;                   // ISA A-frag half select

    v8f acc[2][4] = {};

    for (int kb = 0; kb < KDIM; kb += 32) {
        // --- A fragments: lane m  -> row m, K {kb..kb+7, kb+16..kb+23}
        //                  lane m+16 -> row m, K {kb+8..kb+15, kb+24..kb+31}
        v16h a[2];
        #pragma unroll
        for (int mt = 0; mt < 2; ++mt) {
            const _Float16* ap =
                A + (size_t)(rowBase + mt * 16 + mLane) * KDIM + kb + halfSel * 8;
            union { v16h v; v8h h[2]; } u;
            u.h[0] = *(const v8h*)(ap);              // K block 0 (8 f16, 16B)
            u.h[1] = *(const v8h*)(ap + 16);         // K block +16
            a[mt] = u.v;
        }
        // --- B fragments: lane l -> column N=l%16, K in same order as A fragments
        v16h b[4];
        #pragma unroll
        for (int nt = 0; nt < 4; ++nt) {
            const _Float16* bp = B + (size_t)(kb + halfSel * 8) * OUT_DIMS
                                     + colBase + nt * 16 + mLane;
            #pragma unroll
            for (int s = 0; s < 16; ++s)
                b[nt][s] = bp[(size_t)((s & 7) + (s >> 3) * 16) * OUT_DIMS];
        }
        #pragma unroll
        for (int mt = 0; mt < 2; ++mt) {
            #pragma unroll
            for (int nt = 0; nt < 4; ++nt) {
                acc[mt][nt] = __builtin_amdgcn_wmma_f32_16x16x32_f16(
                    /*neg_a=*/false, a[mt], /*neg_b=*/false, b[nt],
                    /*c_mod=*/(short)0, acc[mt][nt],
                    /*reuse_a=*/false, /*reuse_b=*/false);
            }
        }
    }

    // D layout: VGPR r, lanes 0-15 -> M=r, N=lane; lanes 16-31 -> M=8+r, N=lane-16
    const int rOff = (lane < 16) ? 0 : 8;
    #pragma unroll
    for (int mt = 0; mt < 2; ++mt) {
        #pragma unroll
        for (int nt = 0; nt < 4; ++nt) {
            float* cp = C + (size_t)(rowBase + mt * 16 + rOff) * OUT_DIMS
                          + colBase + nt * 16 + mLane;
            #pragma unroll
            for (int r = 0; r < 8; ++r)
                cp[(size_t)r * OUT_DIMS] = acc[mt][nt][r];
        }
    }
}

// ---------------------------------------------------------------------------
extern "C" void kernel_launch(void* const* d_in, const int* in_sizes, int n_in,
                              void* d_out, int out_size, void* d_ws, size_t ws_size,
                              hipStream_t stream) {
    const float* x      = (const float*)d_in[0];
    const float* coeffs = (const float*)d_in[1];
    float*       out    = (float*)d_out;

    const size_t bBytes = (size_t)KDIM * OUT_DIMS * sizeof(_Float16);   // 4 MB
    _Float16* Bh = (_Float16*)d_ws;
    _Float16* Ah = (_Float16*)((char*)d_ws + bBytes);

    // Pick the largest batch chunk whose f16 activation tile fits in d_ws.
    size_t availA = (ws_size > bBytes) ? (ws_size - bBytes) : 0;
    int chunk = BATCH;
    while (chunk > 128 && (size_t)chunk * KDIM * sizeof(_Float16) > availA)
        chunk >>= 1;

    geg_pack_coeffs<<<(KDIM * OUT_DIMS) / 256, 256, 0, stream>>>(coeffs, Bh);

    for (int b0 = 0; b0 < BATCH; b0 += chunk) {
        geg_poly<<<(chunk * IN_DIMS) / 256, 256, 0, stream>>>(
            x + (size_t)b0 * IN_DIMS, Ah);
        dim3 grid(chunk / 128, OUT_DIMS / 128);
        geg_gemm<<<grid, 256, 0, stream>>>(Ah, Bh, out + (size_t)b0 * OUT_DIMS);
    }
}
